// attn_26955214750309
// MI455X (gfx1250) — matrix operand order repaired and checked by reading, not compiled, not run on hardware
//
#include <hip/hip_runtime.h>

// ---------------------------------------------------------------------------
// Problem constants (from reference): B=8, Tq=1024, Tk=4096, DIN=1024, DOUT=256
// ---------------------------------------------------------------------------
constexpr int B_   = 8;
constexpr int TQ   = 1024;
constexpr int TK   = 4096;
constexpr int DIN  = 1024;
constexpr int DOUT = 256;

constexpr int SROW = TK + 4;   // padded LDS row stride (floats) to avoid 2-way
                               // bank conflicts between the half-wave row sets
constexpr size_t SCORES_LDS_BYTES = (size_t)16 * SROW * sizeof(float); // 256.25KB
constexpr size_t QTILE_LDS_BYTES  = (size_t)16 * DOUT * sizeof(_Float16); // 8KB
constexpr size_t SCORES_DYN_LDS   = SCORES_LDS_BYTES + QTILE_LDS_BYTES;

#define NEG_INF_F (-1e30f)

typedef __attribute__((ext_vector_type(16))) _Float16 v16h;
typedef __attribute__((ext_vector_type(8)))  _Float16 v8h;
typedef __attribute__((ext_vector_type(8)))  float    v8f;
typedef __attribute__((ext_vector_type(4)))  unsigned int u32x4;
typedef __attribute__((ext_vector_type(8)))  int      i32x8;
typedef __attribute__((ext_vector_type(4)))  int      i32x4;

union V16U { v16h v; v8h h[2]; };

// ---------------------------------------------------------------------------
// WMMA fragment loaders (wave32; A/B in this code's contiguous k order, not the ISA table order)
//
// A 16x32 f16 (MxK): lane L holds row M = L&15.
//   lanes 0-15 : elems 0..15 -> K = k0+0..15
//   lanes 16-31: elems 0..15 -> K = k0+16..31
// B 32x16 f16 (KxN): lane L holds column N = L&15; elems e -> K = k0+e+16*hi.
// C/D 16x16 f32: VGPR r -> row M = r + 8*hi, column N = lane&15.
// ---------------------------------------------------------------------------
__device__ __forceinline__ v16h load_a_frag(const _Float16* A, int row0, int k0,
                                            int ld, int lane) {
  int m  = lane & 15;
  int hi = lane >> 4;
  const _Float16* p = A + (size_t)(row0 + m) * ld + (size_t)(k0 + hi * 16);
  V16U u;
  u.h[0] = *(const v8h*)(p);
  u.h[1] = *(const v8h*)(p + 8);
  return u.v;
}

// B where B[k][n] = S[n][k] (S row-major, row stride ld): contiguous per lane.
__device__ __forceinline__ v16h load_b_frag_rowk(const _Float16* S, int n0, int k0,
                                                 int ld, int lane) {
  int n  = lane & 15;
  int hi = lane >> 4;
  const _Float16* p = S + (size_t)(n0 + n) * ld + (size_t)(k0 + hi * 16);
  V16U u;
  u.h[0] = *(const v8h*)(p);
  u.h[1] = *(const v8h*)(p + 8);
  return u.v;
}

__device__ __forceinline__ v8f wmma_f16(v16h a, v16h b, v8f c) {
  // (neg_a, A, neg_b, B, c_mod, C, reuse_a, reuse_b)
  return __builtin_amdgcn_wmma_f32_16x16x32_f16(false, a, false, b, (short)0, c,
                                                false, false);
}

__device__ __forceinline__ v8f zero8() {
  v8f z = {0.f, 0.f, 0.f, 0.f, 0.f, 0.f, 0.f, 0.f};
  return z;
}

// ---------------------------------------------------------------------------
// Tensor Data Mover: 2-D f16 tile (rows x cols) global -> LDS.
// Descriptor packing per CDNA5 ISA 8.3-8.5:
//  group0: [1:0]=count=1, [63:32]=lds_addr, [120:64]=global_addr, [127:126]=2
//  group1: [17:16]=data_size(1=2B), [79:48]=tensor_dim0, [111:80]=tensor_dim1,
//          [127:112]=tile_dim0, [143:128]=tile_dim1, [207:160]=dim0_stride
// ---------------------------------------------------------------------------
#if __has_builtin(__builtin_amdgcn_tensor_load_to_lds)
#define HAVE_TDM 1
__device__ __forceinline__ void tdm_load_2d_f16(unsigned lds_off, const void* gptr,
                                                unsigned rows, unsigned cols,
                                                unsigned stride_elems) {
  unsigned long long ga = (unsigned long long)(size_t)gptr;
  u32x4 g0;
  g0[0] = 1u;                                            // count=1
  g0[1] = lds_off;                                       // lds_addr (bytes)
  g0[2] = (unsigned)(ga & 0xffffffffu);                  // global_addr[31:0]
  g0[3] = (unsigned)((ga >> 32) & 0x1ffffffu) | (2u << 30); // addr[56:32], type=2
  i32x8 g1;
  g1[0] = (int)(1u << 16);                               // data_size = 2 bytes
  g1[1] = (int)((cols & 0xffffu) << 16);                 // tensor_dim0[15:0]
  g1[2] = (int)((cols >> 16) & 0xffffu) | (int)((rows & 0xffffu) << 16); // dim0 hi, dim1 lo
  g1[3] = (int)((rows >> 16) & 0xffffu) | (int)((cols & 0xffffu) << 16); // dim1 hi, tile_dim0
  g1[4] = (int)(rows & 0xffffu);                         // tile_dim1 (tile_dim2=0)
  g1[5] = (int)stride_elems;                             // tensor_dim0_stride[31:0]
  g1[6] = 0;
  g1[7] = 0;
  i32x4 gz = {0, 0, 0, 0};
#if __clang_major__ >= 23
  i32x8 gz8 = {0, 0, 0, 0, 0, 0, 0, 0};
  __builtin_amdgcn_tensor_load_to_lds(g0, g1, gz, gz, gz8, 0);
#else
  __builtin_amdgcn_tensor_load_to_lds(g0, g1, gz, gz, 0);
#endif
}
#else
#define HAVE_TDM 0
#endif

// ---------------------------------------------------------------------------
// Kernel 0a: f32 -> f16 conversion (dec, W)
// ---------------------------------------------------------------------------
__global__ __launch_bounds__(256) void cvt_f32_f16_kernel(const float* __restrict__ src,
                                                          _Float16* __restrict__ dst,
                                                          int n) {
  int i = blockIdx.x * 256 + threadIdx.x;
  if (i < n) dst[i] = (_Float16)src[i];
}

// ---------------------------------------------------------------------------
// Kernel 0b: emb f32 -> emb16 (same layout) + embT16 (transposed [DOUT][Tk])
// ---------------------------------------------------------------------------
__global__ __launch_bounds__(256) void emb_cvt_transpose_kernel(
    const float* __restrict__ emb, _Float16* __restrict__ emb16,
    _Float16* __restrict__ embT16) {
  __shared__ _Float16 tile[16][17];
  int b  = blockIdx.z;
  int k0 = blockIdx.x * 16;
  int o0 = blockIdx.y * 16;
  int tx = threadIdx.x & 15;
  int ty = threadIdx.x >> 4;
  _Float16 h = (_Float16)emb[((size_t)b * TK + k0 + ty) * DOUT + o0 + tx];
  emb16[((size_t)b * TK + k0 + ty) * DOUT + o0 + tx] = h;
  tile[ty][tx] = h;
  __syncthreads();
  embT16[((size_t)b * DOUT + o0 + ty) * TK + k0 + tx] = tile[tx][ty];
}

// ---------------------------------------------------------------------------
// Kernel 1: decsmall[b,q,o] = sum_i dec[b,q,i]*W[o,i] + bias[o]   (f16 out)
// 16x64 output tile per wave (A-fragment reused across 4 WMMAs per k-step).
// ---------------------------------------------------------------------------
__global__ __launch_bounds__(256) void proj_gemm_kernel(const _Float16* __restrict__ dec16,
                                                        const _Float16* __restrict__ w16,
                                                        const float* __restrict__ bias,
                                                        _Float16* __restrict__ ds16) {
  int wave = threadIdx.x >> 5;
  int lane = threadIdx.x & 31;
  int gt   = blockIdx.x * 8 + wave;   // 0..2047 (256 jobs/batch)
  int b    = gt >> 8;
  int rem  = gt & 255;
  int mt   = rem >> 2;
  int ng   = rem & 3;
  int row0 = mt * 16, n0 = ng * 64;

  const _Float16* A = dec16 + (size_t)b * TQ * DIN;
  v8f acc[4] = {zero8(), zero8(), zero8(), zero8()};

  for (int k0 = 0; k0 < DIN; k0 += 32) {
    v16h a = load_a_frag(A, row0, k0, DIN, lane);
#pragma unroll
    for (int j = 0; j < 4; ++j) {
      v16h bb = load_b_frag_rowk(w16, n0 + j * 16, k0, DIN, lane); // B[k][n]=W[n][k]
      acc[j] = wmma_f16(a, bb, acc[j]);
    }
  }

  int cb = lane & 15;
  int hi = lane >> 4;
  _Float16* Dp = ds16 + (size_t)b * TQ * DOUT;
#pragma unroll
  for (int j = 0; j < 4; ++j) {
    int col = n0 + j * 16 + cb;
    float bv = bias[col];
#pragma unroll
    for (int r = 0; r < 8; ++r) {
      Dp[(size_t)(row0 + r + hi * 8) * DOUT + col] = (_Float16)(acc[j][r] + bv);
    }
  }
}

// ---------------------------------------------------------------------------
// Kernel 2: scores + mask + softmax, raw scores held entirely in LDS.
// One block per (batch, 16-query tile); ~270 KB dynamic LDS (CDNA5 WGP has
// 320 KB). TDM stages the shared 16x256 query tile; each wave owns a 512-key
// strip and keeps its 16x512 f32 score strip in LDS, so the only HBM traffic
// is the final normalized f32 + f16 attention writes.
// ---------------------------------------------------------------------------
__global__ __launch_bounds__(256) void scores_softmax_kernel(
    const _Float16* __restrict__ ds16, const _Float16* __restrict__ emb16,
    const int* __restrict__ emask, float* __restrict__ attn,
    _Float16* __restrict__ attn16) {
  extern __shared__ char smem_raw[];
  float*    sc    = (float*)smem_raw;                       // [16][SROW]
  _Float16* qtile = (_Float16*)(smem_raw + SCORES_LDS_BYTES); // [16*DOUT]

  __shared__ float lm[8][16];
  __shared__ float ls[8][16];
  __shared__ float fm[16];
  __shared__ float fs[16];

  int wave = threadIdx.x >> 5;
  int lane = threadIdx.x & 31;
  int hi   = lane >> 4;
  int b    = blockIdx.x >> 6;
  int q0   = (blockIdx.x & 63) * 16;

  const _Float16* Aq = ds16 + ((size_t)b * TQ + q0) * DOUT;   // 16 x 256
  const _Float16* Eb = emb16 + (size_t)b * TK * DOUT;          // 4096 x 256
  const int* Mb = emask + (size_t)b * TK;
  float* Pf = attn + ((size_t)b * TQ + q0) * TK;
  _Float16* Ph = attn16 + ((size_t)b * TQ + q0) * TK;

#if HAVE_TDM
  if (threadIdx.x == 0) {   // wave 0 issues the DMA (TDM ignores EXEC)
    tdm_load_2d_f16((unsigned)(size_t)(void*)qtile, Aq, 16u, (unsigned)DOUT,
                    (unsigned)DOUT);
    __builtin_amdgcn_s_wait_tensorcnt(0);
  }
#else
  // 16x256 tile is contiguous (row stride == DOUT): cooperative flat copy.
  for (int i = threadIdx.x; i < 16 * DOUT; i += 256) qtile[i] = Aq[i];
#endif
  __syncthreads();

  // Query-tile A fragments from LDS: loaded once, reused for all 32 key tiles.
  v16h afrag[8];
#pragma unroll
  for (int kk = 0; kk < 8; ++kk)
    afrag[kk] = load_a_frag(qtile, 0, kk * 32, DOUT, lane);

  float m[8], s[8];
#pragma unroll
  for (int r = 0; r < 8; ++r) { m[r] = -3.0e38f; s[r] = 0.0f; }

  for (int t = 0; t < 32; ++t) {
    int n0 = wave * 512 + t * 16;
    v8f acc = zero8();
#pragma unroll
    for (int kk = 0; kk < 8; ++kk) {
      // B[k=o][n=key] = emb[key][o] : contiguous per lane
      v16h bfrag = load_b_frag_rowk(Eb, n0, kk * 32, DOUT, lane);
      acc = wmma_f16(afrag[kk], bfrag, acc);
    }
    int col = n0 + (lane & 15);
    bool ok = (Mb[col] != 0);
#pragma unroll
    for (int r = 0; r < 8; ++r) {
      float x = ok ? acc[r] : NEG_INF_F;
      sc[(size_t)(r + hi * 8) * SROW + col] = x;   // raw masked score -> LDS
      if (x > m[r]) { s[r] = s[r] * __expf(m[r] - x) + 1.0f; m[r] = x; }
      else          { s[r] += __expf(x - m[r]); }
    }
  }

  // Reduce (m,s) across the 16 lanes of each half-wave (rows differ per half).
#pragma unroll
  for (int off = 1; off < 16; off <<= 1) {
#pragma unroll
    for (int r = 0; r < 8; ++r) {
      float mo = __shfl_xor(m[r], off, 32);
      float so = __shfl_xor(s[r], off, 32);
      float mn = fmaxf(m[r], mo);
      s[r] = s[r] * __expf(m[r] - mn) + so * __expf(mo - mn);
      m[r] = mn;
    }
  }
  if ((lane & 15) == 0) {
#pragma unroll
    for (int r = 0; r < 8; ++r) { lm[wave][hi * 8 + r] = m[r]; ls[wave][hi * 8 + r] = s[r]; }
  }
  __syncthreads();

  if (threadIdx.x < 16) {
    float M = -3.0e38f, S = 0.0f;
#pragma unroll
    for (int w = 0; w < 8; ++w) {
      float mw = lm[w][threadIdx.x], sw = ls[w][threadIdx.x];
      float mn = fmaxf(M, mw);
      S = S * __expf(M - mn) + sw * __expf(mw - mn);
      M = mn;
    }
    fm[threadIdx.x] = M;
    fs[threadIdx.x] = (S > 0.0f) ? S : 1.0f;
  }
  __syncthreads();

  // Pass 2: normalize straight out of LDS; only normalized results hit HBM.
  for (int t = 0; t < 32; ++t) {
    int col = wave * 512 + t * 16 + (lane & 15);
#pragma unroll
    for (int r = 0; r < 8; ++r) {
      int row = r + hi * 8;
      float x = sc[(size_t)row * SROW + col];
      float p = __expf(x - fm[row]) / fs[row];  // masked: exp(-huge) -> 0
      size_t idx = (size_t)row * TK + col;
      Pf[idx] = p;
      Ph[idx] = (_Float16)p;
    }
  }
}

// ---------------------------------------------------------------------------
// Kernel 3: out[b,q,o] = sum_k attn[b,q,k] * emb[b,k,o]   (f32 out)
// 16x64 tile per wave; B fragments from transposed embT16 (contiguous loads).
// ---------------------------------------------------------------------------
__global__ __launch_bounds__(256) void out_gemm_kernel(const _Float16* __restrict__ attn16,
                                                       const _Float16* __restrict__ embT16,
                                                       float* __restrict__ out) {
  int wave = threadIdx.x >> 5;
  int lane = threadIdx.x & 31;
  int gt   = blockIdx.x * 8 + wave;   // 0..2047
  int b    = gt >> 8;
  int rem  = gt & 255;
  int mt   = rem >> 2;
  int ng   = rem & 3;
  int row0 = mt * 16, n0 = ng * 64;

  const _Float16* A  = attn16 + (size_t)b * TQ * TK;    // 1024 x 4096
  const _Float16* ET = embT16 + (size_t)b * DOUT * TK;  // 256 x 4096 (= emb^T)
  v8f acc[4] = {zero8(), zero8(), zero8(), zero8()};

  for (int k0 = 0; k0 < TK; k0 += 32) {
    v16h a = load_a_frag(A, row0, k0, TK, lane);
    __builtin_prefetch(A + (size_t)(row0 + (lane & 15)) * TK + k0 + 64, 0, 1);
#pragma unroll
    for (int j = 0; j < 4; ++j) {
      // B[k][n] = emb[k][n] = embT[n][k] : contiguous per lane
      v16h bb = load_b_frag_rowk(ET, n0 + j * 16, k0, TK, lane);
      acc[j] = wmma_f16(a, bb, acc[j]);
    }
  }

  int cb = lane & 15;
  int hi = lane >> 4;
  float* Op = out + (size_t)b * TQ * DOUT;
#pragma unroll
  for (int j = 0; j < 4; ++j) {
    int col = n0 + j * 16 + cb;
#pragma unroll
    for (int r = 0; r < 8; ++r) {
      Op[(size_t)(row0 + r + hi * 8) * DOUT + col] = acc[j][r];
    }
  }
}

// ---------------------------------------------------------------------------
// Host-side launch
// ---------------------------------------------------------------------------
extern "C" void kernel_launch(void* const* d_in, const int* in_sizes, int n_in,
                              void* d_out, int out_size, void* d_ws, size_t ws_size,
                              hipStream_t stream) {
  (void)in_sizes; (void)n_in; (void)out_size; (void)ws_size;

  const float* dec   = (const float*)d_in[0];   // [B,Tq,DIN]   f32
  const float* emb   = (const float*)d_in[1];   // [B,Tk,DOUT]  f32
  const int*   emask = (const int*)d_in[2];     // [B,Tk]       i32
  const float* W     = (const float*)d_in[3];   // [DOUT,DIN]   f32
  const float* bias  = (const float*)d_in[4];   // [DOUT]       f32

  float* out  = (float*)d_out;                          // [B,Tq,DOUT]
  float* attn = out + (size_t)B_ * TQ * DOUT;           // [B,Tq,Tk]

  // Workspace layout (f16 staging, ~116.5 MB total)
  _Float16* dec16  = (_Float16*)d_ws;
  _Float16* emb16  = dec16  + (size_t)B_ * TQ * DIN;
  _Float16* embT16 = emb16  + (size_t)B_ * TK * DOUT;
  _Float16* w16    = embT16 + (size_t)B_ * TK * DOUT;
  _Float16* ds16   = w16    + (size_t)DOUT * DIN;
  _Float16* attn16 = ds16   + (size_t)B_ * TQ * DOUT;

  // Allow > 64 KB dynamic LDS for the softmax kernel (CDNA5 WGP: 320 KB).
  // Immediate (non-stream) runtime call: safe under graph capture.
  hipFuncSetAttribute((const void*)scores_softmax_kernel,
                      hipFuncAttributeMaxDynamicSharedMemorySize,
                      (int)SCORES_DYN_LDS);

  int n1 = B_ * TQ * DIN;
  int n3 = DOUT * DIN;
  cvt_f32_f16_kernel<<<(n1 + 255) / 256, 256, 0, stream>>>(dec, dec16, n1);
  cvt_f32_f16_kernel<<<(n3 + 255) / 256, 256, 0, stream>>>(W, w16, n3);
  emb_cvt_transpose_kernel<<<dim3(TK / 16, DOUT / 16, B_), 256, 0, stream>>>(
      emb, emb16, embT16);

  // 16x64 tile per wave, 8 waves/block: 8*64*4/8 = 256 blocks
  proj_gemm_kernel<<<256, 256, 0, stream>>>(dec16, w16, bias, ds16);

  // one block per (batch, 16-query tile): 8*64 = 512 blocks
  scores_softmax_kernel<<<B_ * (TQ / 16), 256, SCORES_DYN_LDS, stream>>>(
      ds16, emb16, emask, attn, attn16);

  out_gemm_kernel<<<256, 256, 0, stream>>>(attn16, embT16, out);
}
